// EncoderLayer_56453050138711
// MI455X (gfx1250) — compile-verified, matrix operand order checked by reading, not run on hardware
//
#include <hip/hip_runtime.h>
#include <hip/hip_bf16.h>
#include <cstdint>

typedef __attribute__((ext_vector_type(16))) _Float16 v16h;
typedef __attribute__((ext_vector_type(8)))  _Float16 v8h;
typedef __attribute__((ext_vector_type(8)))  float    v8f;

#define S_LEN   1500
#define S_PAD   1536      // multiple of 128 (wg M-tile) and 32 (wmma K-step)
#define NBATCH  16
#define DMODEL  512
#define DFF     2048

// ---------------------------------------------------------------------------
// WMMA A/B fragment load from a 32-wide row-major LDS tile (ISA 7.12.2):
// lane<16 -> k-half 0, lane>=16 -> k-half 8; elements 0..7 hold k=kh..kh+7,
// elements 8..15 hold k=16+kh..16+kh+7 (two contiguous 16B ds loads).
// ---------------------------------------------------------------------------
__device__ __forceinline__ v16h load_frag_lds(const _Float16* tile, int r0) {
  const int lane = threadIdx.x & 31;
  const int r    = r0 + (lane & 15);
  const int kh   = (lane >> 4) * 8;
  const _Float16* p = tile + r * 32 + kh;
  v8h lo = *(const v8h*)(p);        // ds_load_b128
  v8h hi = *(const v8h*)(p + 16);   // ds_load_b128
  v16h f;
#pragma unroll
  for (int i = 0; i < 8; ++i) { f[i] = lo[i]; f[i + 8] = hi[i]; }
  return f;
}

// ---------------------------------------------------------------------------
// Generic WMMA GEMM: C[M,N] = A[M,K] * W[N,K]^T  (+ epilogue)
// Block = 128 threads (4 waves). Workgroup tile 128x64; each wave owns a
// 32x64 C tile (2 A-fragments x 4 B-fragments = 8 wmma per K-step, 64 acc
// VGPRs). Global->LDS staging via gfx1250 async-to-LDS copies (ASYNCcnt),
// double buffered so the next K-step's tiles stream in during the WMMAs.
// EPI: 0 = store f16          1 = store f16 transposed per-batch (vT)
//      2 = *alpha, store f32  3 = +residual(unpadded f32), store f32
//      4 = +bias, SiLU, f16   5 = +bias +residual(padded f32), store f32
// ---------------------------------------------------------------------------
template <int EPI>
__global__ void __launch_bounds__(128)
gemm_wmma(const _Float16* __restrict__ A, int lda,
          const _Float16* __restrict__ W, int ldw,
          void* __restrict__ out, int ldo,
          const float* __restrict__ bias,
          const float* __restrict__ res,
          int K, float alpha) {
  __shared__ __align__(16) _Float16 sA[2][128 * 32];  // 16KB
  __shared__ __align__(16) _Float16 sB[2][64 * 32];   //  8KB
  const int tid  = threadIdx.x;
  const int wave = tid >> 5;
  const int lane = tid & 31;
  const int rowT = blockIdx.y * 128;
  const int colT = blockIdx.x * 64;

  v8f acc[2][4];
#pragma unroll
  for (int r = 0; r < 2; ++r)
#pragma unroll
    for (int j = 0; j < 4; ++j)
      acc[r][j] = (v8f){0.f,0.f,0.f,0.f,0.f,0.f,0.f,0.f};

  // Stage the 128x32 A tile (512 16B chunks) and 64x32 W tile (256 chunks)
  // into LDS buffer `buf` for k0 via per-lane async global->LDS copies
  // (generic LDS address low 32 bits == LDS offset, ISA 10.2).
  auto issue = [&](int buf, int k0) {
#pragma unroll
    for (int c = tid; c < 512; c += 128) {
      const int r = c >> 2, c4 = c & 3;
      const _Float16* ga = A + (size_t)(rowT + r) * lda + k0 + c4 * 8;
      unsigned la = (unsigned)(uintptr_t)&sA[buf][r * 32 + c4 * 8];
      asm volatile("global_load_async_to_lds_b128 %0, %1, off"
                   :: "v"(la), "v"(ga) : "memory");
    }
#pragma unroll
    for (int c = tid; c < 256; c += 128) {
      const int r = c >> 2, c4 = c & 3;
      const _Float16* gb = W + (size_t)(colT + r) * ldw + k0 + c4 * 8;
      unsigned lb = (unsigned)(uintptr_t)&sB[buf][r * 32 + c4 * 8];
      asm volatile("global_load_async_to_lds_b128 %0, %1, off"
                   :: "v"(lb), "v"(gb) : "memory");
    }
  };

  issue(0, 0);
  int buf = 0;
  for (int k0 = 0; k0 < K; k0 += 32, buf ^= 1) {
    asm volatile("s_wait_asynccnt 0x0" ::: "memory");  // my async copies landed
    __syncthreads();                                   // everyone's landed; prev reads done
    if (k0 + 32 < K) issue(buf ^ 1, k0 + 32);          // overlap next stage with compute

    v16h a0 = load_frag_lds(&sA[buf][0], wave * 32);
    v16h a1 = load_frag_lds(&sA[buf][0], wave * 32 + 16);
#pragma unroll
    for (int j = 0; j < 4; ++j) {
      v16h bf = load_frag_lds(&sB[buf][0], j * 16);
      acc[0][j] = __builtin_amdgcn_wmma_f32_16x16x32_f16(
          false, a0, false, bf, (short)0, acc[0][j], false, false);
      acc[1][j] = __builtin_amdgcn_wmma_f32_16x16x32_f16(
          false, a1, false, bf, (short)0, acc[1][j], false, false);
    }
  }

  // C layout (ISA 7.12.2): lane<16 -> rows row0+i, lane>=16 -> rows row0+8+i;
  // n = col0 + (lane&15) for both halves.
  const int nb = colT + (lane & 15);
#pragma unroll
  for (int r = 0; r < 2; ++r) {
    const int mb = rowT + wave * 32 + r * 16 + ((lane >> 4) << 3);
#pragma unroll
    for (int j = 0; j < 4; ++j) {
      const int n = nb + j * 16;
#pragma unroll
      for (int i = 0; i < 8; ++i) {
        const int m = mb + i;
        float v = acc[r][j][i];
        if constexpr (EPI == 0) {
          ((_Float16*)out)[(size_t)m * ldo + n] = (_Float16)v;
        } else if constexpr (EPI == 1) {
          const int b = m / S_PAD, s = m % S_PAD;   // vT[b][n][s]
          ((_Float16*)out)[((size_t)b * DMODEL + n) * S_PAD + s] = (_Float16)v;
        } else if constexpr (EPI == 2) {
          ((float*)out)[(size_t)m * ldo + n] = v * alpha;
        } else if constexpr (EPI == 3) {
          const int b = m / S_PAD, s = m % S_PAD;
          if (s < S_LEN)
            ((float*)out)[(size_t)m * ldo + n] =
                v + res[((size_t)b * S_LEN + s) * DMODEL + n];
        } else if constexpr (EPI == 4) {
          v += bias[n];
          v = v / (1.f + __expf(-v));               // SiLU
          ((_Float16*)out)[(size_t)m * ldo + n] = (_Float16)v;
        } else if constexpr (EPI == 5) {
          v += bias[n] + res[(size_t)m * ldo + n];
          ((float*)out)[(size_t)m * ldo + n] = v;
        }
      }
    }
  }
}

// ---------------------------------------------------------------------------
// LayerNorm (one 256-thread block per padded row of 512).
// ---------------------------------------------------------------------------
__global__ void ln1_kernel(const float* __restrict__ y,
                           const float* __restrict__ g,
                           const float* __restrict__ be,
                           float* __restrict__ xo,          // padded f32
                           _Float16* __restrict__ xh) {     // padded f16
  const int row = blockIdx.x, tid = threadIdx.x;
  const int s = row % S_PAD;
  __shared__ float red[256];
  if (s >= S_LEN) {
    for (int d = tid; d < DMODEL; d += 256) {
      xo[(size_t)row * DMODEL + d] = 0.f;
      xh[(size_t)row * DMODEL + d] = (_Float16)0.f;
    }
    return;
  }
  const float* yr = y + (size_t)row * DMODEL;
  float lsum = 0.f;
  for (int d = tid; d < DMODEL; d += 256) lsum += yr[d];
  red[tid] = lsum; __syncthreads();
  for (int o = 128; o > 0; o >>= 1) { if (tid < o) red[tid] += red[tid + o]; __syncthreads(); }
  const float mu = red[0] * (1.f / DMODEL); __syncthreads();
  float lvar = 0.f;
  for (int d = tid; d < DMODEL; d += 256) { float t = yr[d] - mu; lvar += t * t; }
  red[tid] = lvar; __syncthreads();
  for (int o = 128; o > 0; o >>= 1) { if (tid < o) red[tid] += red[tid + o]; __syncthreads(); }
  const float rstd = rsqrtf(red[0] * (1.f / DMODEL) + 1e-4f);
  for (int d = tid; d < DMODEL; d += 256) {
    float v = (yr[d] - mu) * rstd * g[d] + be[d];
    xo[(size_t)row * DMODEL + d] = v;
    xh[(size_t)row * DMODEL + d] = (_Float16)v;
  }
}

__global__ void ln2_kernel(const float* __restrict__ y,
                           const float* __restrict__ g,
                           const float* __restrict__ be,
                           float* __restrict__ outp) {      // unpadded [B,S,D]
  const int row = blockIdx.x, tid = threadIdx.x;
  const int s = row % S_PAD, b = row / S_PAD;
  if (s >= S_LEN) return;
  __shared__ float red[256];
  const float* yr = y + (size_t)row * DMODEL;
  float lsum = 0.f;
  for (int d = tid; d < DMODEL; d += 256) lsum += yr[d];
  red[tid] = lsum; __syncthreads();
  for (int o = 128; o > 0; o >>= 1) { if (tid < o) red[tid] += red[tid + o]; __syncthreads(); }
  const float mu = red[0] * (1.f / DMODEL); __syncthreads();
  float lvar = 0.f;
  for (int d = tid; d < DMODEL; d += 256) { float t = yr[d] - mu; lvar += t * t; }
  red[tid] = lvar; __syncthreads();
  for (int o = 128; o > 0; o >>= 1) { if (tid < o) red[tid] += red[tid + o]; __syncthreads(); }
  const float rstd = rsqrtf(red[0] * (1.f / DMODEL) + 1e-4f);
  float* orow = outp + ((size_t)b * S_LEN + s) * DMODEL;
  for (int d = tid; d < DMODEL; d += 256)
    orow[d] = (yr[d] - mu) * rstd * g[d] + be[d];
}

// ---------------------------------------------------------------------------
// Softmax over one scores row (valid cols < S_LEN); writes f16 probs,
// zero-fills padded cols and padded rows so P.V contributes nothing there.
// ---------------------------------------------------------------------------
__global__ void softmax_kernel(const float* __restrict__ sc,
                               _Float16* __restrict__ p) {
  const int row = blockIdx.x, tid = threadIdx.x;
  _Float16* pr = p + (size_t)row * S_PAD;
  if (row >= S_LEN) {
    for (int c = tid; c < S_PAD; c += 256) pr[c] = (_Float16)0.f;
    return;
  }
  __shared__ float red[256];
  const float* sr = sc + (size_t)row * S_PAD;
  float lmax = -3.4e38f;
  for (int c = tid; c < S_LEN; c += 256) lmax = fmaxf(lmax, sr[c]);
  red[tid] = lmax; __syncthreads();
  for (int o = 128; o > 0; o >>= 1) { if (tid < o) red[tid] = fmaxf(red[tid], red[tid + o]); __syncthreads(); }
  const float mx = red[0]; __syncthreads();
  float lsum = 0.f;
  for (int c = tid; c < S_LEN; c += 256) lsum += __expf(sr[c] - mx);
  red[tid] = lsum; __syncthreads();
  for (int o = 128; o > 0; o >>= 1) { if (tid < o) red[tid] += red[tid + o]; __syncthreads(); }
  const float inv = 1.f / red[0];
  for (int c = tid; c < S_PAD; c += 256)
    pr[c] = (c < S_LEN) ? (_Float16)(__expf(sr[c] - mx) * inv) : (_Float16)0.f;
}

// ---------------------------------------------------------------------------
// f32 -> f16 conversions
// ---------------------------------------------------------------------------
__global__ void cvt_f32_f16(const float* __restrict__ s, _Float16* __restrict__ d, int n) {
  int i = blockIdx.x * 256 + threadIdx.x;
  if (i < n) d[i] = (_Float16)s[i];
}

__global__ void cvt_pad_x(const float* __restrict__ x, _Float16* __restrict__ xh) {
  size_t i = (size_t)blockIdx.x * 256 + threadIdx.x;
  const size_t total = (size_t)NBATCH * S_PAD * DMODEL;
  if (i >= total) return;
  int d = (int)(i % DMODEL);
  size_t r = i / DMODEL;
  int s = (int)(r % S_PAD), b = (int)(r / S_PAD);
  xh[i] = (s < S_LEN) ? (_Float16)x[((size_t)b * S_LEN + s) * DMODEL + d]
                      : (_Float16)0.f;
}

// ---------------------------------------------------------------------------
extern "C" void kernel_launch(void* const* d_in, const int* in_sizes, int n_in,
                              void* d_out, int out_size, void* d_ws, size_t ws_size,
                              hipStream_t stream) {
  (void)in_sizes; (void)n_in; (void)out_size; (void)ws_size;
  const float* x     = (const float*)d_in[0];
  const float* Wq    = (const float*)d_in[1];
  const float* Wk    = (const float*)d_in[2];
  const float* Wv    = (const float*)d_in[3];
  const float* Wo    = (const float*)d_in[4];
  const float* ln1_g = (const float*)d_in[5];
  const float* ln1_b = (const float*)d_in[6];
  const float* W1    = (const float*)d_in[7];
  const float* b1    = (const float*)d_in[8];
  const float* W2    = (const float*)d_in[9];
  const float* b2    = (const float*)d_in[10];
  const float* ln2_g = (const float*)d_in[11];
  const float* ln2_b = (const float*)d_in[12];
  float* outp = (float*)d_out;

  const size_t MP = (size_t)NBATCH * S_PAD;            // 24576 padded rows
  char* w = (char*)d_ws;
  size_t off = 0;
  auto alloc = [&](size_t bytes) -> void* {
    void* p = w + off; off += (bytes + 255) & ~(size_t)255; return p;
  };
  _Float16* xh     = (_Float16*)alloc(MP * DMODEL * 2);   // x f16 (reused as x1h)
  _Float16* qh     = (_Float16*)alloc(MP * DMODEL * 2);
  _Float16* kh     = (_Float16*)alloc(MP * DMODEL * 2);
  _Float16* vT     = (_Float16*)alloc(MP * DMODEL * 2);   // [B][D][S_PAD]
  _Float16* aoh    = (_Float16*)alloc(MP * DMODEL * 2);
  _Float16* hbuf   = (_Float16*)alloc(MP * DFF * 2);
  float*    scores = (float*)   alloc((size_t)S_PAD * S_PAD * 4);   // per-batch reuse
  _Float16* probs  = (_Float16*)alloc((size_t)S_PAD * S_PAD * 2);   // per-batch reuse
  float*    y1     = (float*)   alloc(MP * DMODEL * 4);   // reused as y2
  float*    x1     = (float*)   alloc(MP * DMODEL * 4);
  _Float16* Wqh    = (_Float16*)alloc((size_t)DMODEL * DMODEL * 2);
  _Float16* Wkh    = (_Float16*)alloc((size_t)DMODEL * DMODEL * 2);
  _Float16* Wvh    = (_Float16*)alloc((size_t)DMODEL * DMODEL * 2);
  _Float16* Woh    = (_Float16*)alloc((size_t)DMODEL * DMODEL * 2);
  _Float16* W1h    = (_Float16*)alloc((size_t)DFF * DMODEL * 2);
  _Float16* W2h    = (_Float16*)alloc((size_t)DMODEL * DFF * 2);
  _Float16* x1h    = xh;                                   // alias (x f16 dead)
  float*    y2     = y1;                                   // alias (y1 dead post-LN1)

  const dim3 blk(128);
  const float scale = 0.044194173824159216f;               // 1/sqrt(512)

  // --- precision conversion -------------------------------------------------
  cvt_pad_x<<<(unsigned)((MP * DMODEL + 255) / 256), 256, 0, stream>>>(x, xh);
  cvt_f32_f16<<<(DMODEL*DMODEL+255)/256, 256, 0, stream>>>(Wq, Wqh, DMODEL*DMODEL);
  cvt_f32_f16<<<(DMODEL*DMODEL+255)/256, 256, 0, stream>>>(Wk, Wkh, DMODEL*DMODEL);
  cvt_f32_f16<<<(DMODEL*DMODEL+255)/256, 256, 0, stream>>>(Wv, Wvh, DMODEL*DMODEL);
  cvt_f32_f16<<<(DMODEL*DMODEL+255)/256, 256, 0, stream>>>(Wo, Woh, DMODEL*DMODEL);
  cvt_f32_f16<<<(DFF*DMODEL+255)/256, 256, 0, stream>>>(W1, W1h, DFF*DMODEL);
  cvt_f32_f16<<<(DMODEL*DFF+255)/256, 256, 0, stream>>>(W2, W2h, DMODEL*DFF);

  // --- Q/K/V projections (M=24576, N=512, K=512) ---------------------------
  {
    dim3 g(DMODEL / 64, (unsigned)(MP / 128));
    gemm_wmma<0><<<g, blk, 0, stream>>>(xh, DMODEL, Wqh, DMODEL, qh, DMODEL, nullptr, nullptr, DMODEL, 1.f);
    gemm_wmma<0><<<g, blk, 0, stream>>>(xh, DMODEL, Wkh, DMODEL, kh, DMODEL, nullptr, nullptr, DMODEL, 1.f);
    gemm_wmma<1><<<g, blk, 0, stream>>>(xh, DMODEL, Wvh, DMODEL, vT, S_PAD,  nullptr, nullptr, DMODEL, 1.f);
  }

  // --- attention, one batch at a time (scores/probs scratch reused) --------
  for (int b = 0; b < NBATCH; ++b) {
    const _Float16* qb = qh + (size_t)b * S_PAD * DMODEL;
    const _Float16* kb = kh + (size_t)b * S_PAD * DMODEL;
    const _Float16* vb = vT + (size_t)b * (size_t)DMODEL * S_PAD;
    _Float16* ab = aoh + (size_t)b * S_PAD * DMODEL;
    dim3 gs(S_PAD / 64, S_PAD / 128);
    gemm_wmma<2><<<gs, blk, 0, stream>>>(qb, DMODEL, kb, DMODEL, scores, S_PAD, nullptr, nullptr, DMODEL, scale);
    softmax_kernel<<<S_PAD, 256, 0, stream>>>(scores, probs);
    dim3 gv(DMODEL / 64, S_PAD / 128);
    gemm_wmma<0><<<gv, blk, 0, stream>>>(probs, S_PAD, vb, S_PAD, ab, DMODEL, nullptr, nullptr, S_PAD, 1.f);
  }

  // --- Wo projection + residual, LN1 ---------------------------------------
  {
    dim3 g(DMODEL / 64, (unsigned)(MP / 128));
    gemm_wmma<3><<<g, blk, 0, stream>>>(aoh, DMODEL, Woh, DMODEL, y1, DMODEL, nullptr, x, DMODEL, 1.f);
  }
  ln1_kernel<<<(unsigned)MP, 256, 0, stream>>>(y1, ln1_g, ln1_b, x1, x1h);

  // --- FFN: h = SiLU(x1 W1^T + b1); y2 = h W2^T + b2 + x1; LN2 -------------
  {
    dim3 g1(DFF / 64, (unsigned)(MP / 128));
    gemm_wmma<4><<<g1, blk, 0, stream>>>(x1h, DMODEL, W1h, DMODEL, hbuf, DFF, b1, nullptr, DMODEL, 1.f);
    dim3 g2(DMODEL / 64, (unsigned)(MP / 128));
    gemm_wmma<5><<<g2, blk, 0, stream>>>(hbuf, DFF, W2h, DFF, y2, DMODEL, b2, x1, DFF, 1.f);
  }
  ln2_kernel<<<(unsigned)MP, 256, 0, stream>>>(y2, ln2_g, ln2_b, outp);
}
